// MultiHeadAttention_14963666059964
// MI455X (gfx1250) — matrix operand order repaired and checked by reading, not compiled, not run on hardware
//
#include <hip/hip_runtime.h>

// ---------------------------------------------------------------------------
// Problem constants (match reference)
// ---------------------------------------------------------------------------
#define DIMSZ 512
#define HEADS 8
#define HD    64
#define BSZ   8
#define MSZ   1024
#define NSZ   1024
#define ROWS  (BSZ * MSZ)          // 8192 flattened (b, m) rows

typedef __attribute__((ext_vector_type(16))) __bf16 v16bf;
typedef __attribute__((ext_vector_type(8)))  float  v8f;

union Frag16 {
    v16bf v;
    unsigned short u[16];
    uint4 q[2];
};

// f32 -> bf16 via hardware fptrunc (backend picks v_cvt_*bf16* forms)
static __device__ inline unsigned short f2bf(float f) {
    union { __bf16 h; unsigned short s; } t;
    t.h = (__bf16)f;
    return t.s;
}

// pack 4 floats -> 4 bf16, one 8-byte store
static __device__ inline void st4bf(unsigned short* p, float a, float b, float c, float d) {
    union { unsigned short s[4]; uint2 v; } t;
    t.s[0] = f2bf(a); t.s[1] = f2bf(b); t.s[2] = f2bf(c); t.s[3] = f2bf(d);
    *(uint2*)p = t.v;
}

// A-fragment (16x32 bf16): lane row = lane&15; per-lane elements live at
// k = hf*8..hf*8+7 and 16+hf*8..16+hf*8+7 -> two 16B LDS loads from a
// row-major (stride 32) tile row.
static __device__ inline void ldfragA(Frag16& f, const unsigned short* row, int hf) {
    f.q[0] = *(const uint4*)(row + hf * 8);
    f.q[1] = *(const uint4*)(row + 16 + hf * 8);
}
// B-fragment: per-lane elements are k = hf*8..hf*8+7 and 16+hf*8..16+hf*8+7
// for one column -> two 16B loads from a column-major (stride 32) tile.
static __device__ inline void ldfragB(Frag16& f, const unsigned short* colbase, int hf) {
    f.q[0] = *(const uint4*)(colbase + hf * 8);
    f.q[1] = *(const uint4*)(colbase + 16 + hf * 8);
}

static __device__ inline v8f wmma_bf16(v16bf a, v16bf b, v8f c) {
    return __builtin_amdgcn_wmma_f32_16x16x32_bf16(
        false, a, false, b, (short)0, c, false, false);
}

// ---------------------------------------------------------------------------
// Kernel 0: pack mask[1024][1024] (int 0/1) into bit-mask, 16 bits per u16.
// Mp[m*64 + n/16] bit (n%16)  ==  mask[m][n]
// ---------------------------------------------------------------------------
__global__ __launch_bounds__(256) void pack_mask_kernel(
    const int* __restrict__ mask, unsigned short* __restrict__ Mp)
{
    int w = blockIdx.x * 256 + threadIdx.x;          // 0 .. 65535
    int m = w >> 6, ng = w & 63;
    const int* row = mask + (size_t)m * NSZ + ng * 16;
    unsigned bits = 0;
    #pragma unroll
    for (int u = 0; u < 16; ++u) bits |= (unsigned)(row[u] & 1) << u;
    Mp[w] = (unsigned short)bits;
}

// ---------------------------------------------------------------------------
// Kernel 1: head projections (r = x@l1+bl, s = y@l2, t = x@t1+bt).
// Block: 128 rows x 64 cols for one (head, p). Output bf16 [b][k][row][64].
// ---------------------------------------------------------------------------
__global__ __launch_bounds__(256) void proj_kernel(
    const float* __restrict__ x, const float* __restrict__ y,
    const float* __restrict__ l1, const float* __restrict__ l2,
    const float* __restrict__ t1,
    const float* __restrict__ bl, const float* __restrict__ bt,
    unsigned short* __restrict__ wr, unsigned short* __restrict__ ws,
    unsigned short* __restrict__ wt)
{
    const int rb = blockIdx.x;
    const int kh = blockIdx.y;
    const int p  = blockIdx.z;

    const float* src  = (p == 1) ? y : x;
    const float* W    = (p == 0) ? (l1 + (size_t)kh * DIMSZ * HD)
                      : (p == 1) ? (l2 + (size_t)kh * DIMSZ * HD)
                                 : (t1 + (size_t)kh * DIMSZ * HD);
    const float* bias = (p == 0) ? (bl + kh * HD)
                      : (p == 2) ? (bt + kh * HD) : nullptr;
    unsigned short* dst = (p == 0) ? wr : (p == 1) ? ws : wt;

    __shared__ alignas(16) unsigned short Ab[128 * 32]; // activations, row-major
    __shared__ alignas(16) unsigned short Bt[64 * 32];  // weights, col-major [n][k]

    const int tid  = threadIdx.x;
    const int wave = tid >> 5;
    const int lane = tid & 31;
    const int hf   = lane >> 4;
    const int l15  = lane & 15;
    const int row0 = rb * 128;
    const int wr0  = wave * 16;

    v8f acc[4] = {};

    for (int c0 = 0; c0 < DIMSZ; c0 += 32) {
        __syncthreads();
        // A: 128x32, float4 global load -> packed bf16 (4 groups/thread)
        #pragma unroll
        for (int it = 0; it < 4; ++it) {
            int idx = (tid + it * 256) * 4;
            int rr = idx >> 5, cc = idx & 31;
            float4 f = *(const float4*)&src[(size_t)(row0 + rr) * DIMSZ + c0 + cc];
            st4bf(&Ab[idx], f.x, f.y, f.z, f.w);
        }
        // B transposed: Bt[n][k] <- W[c0+k][n]; 4 scalar loads + 1 b64 store
        #pragma unroll
        for (int it = 0; it < 2; ++it) {
            int g = tid + it * 256;
            int n = g & 63, kq = g >> 6;
            const float* wp = &W[(size_t)(c0 + kq * 4) * HD + n];
            st4bf(&Bt[n * 32 + kq * 4], wp[0], wp[HD], wp[2 * HD], wp[3 * HD]);
        }
        if (c0 + 32 < DIMSZ)
            __builtin_prefetch(&src[(size_t)(row0 + (tid >> 1)) * DIMSZ + c0 + 32], 0, 0);
        __syncthreads();

        Frag16 a;
        ldfragA(a, &Ab[(wr0 + l15) * 32], hf);
        #pragma unroll
        for (int t = 0; t < 4; ++t) {
            Frag16 b;
            ldfragB(b, &Bt[(t * 16 + l15) * 32], hf);
            acc[t] = wmma_bf16(a.v, b.v, acc[t]);
        }
    }

    #pragma unroll
    for (int t = 0; t < 4; ++t) {
        #pragma unroll
        for (int j = 0; j < 8; ++j) {
            int g  = row0 + wr0 + j + 8 * hf;
            int b  = g >> 10, rl = g & 1023, d = t * 16 + l15;
            float v = acc[t][j];
            if (bias) v += bias[d];
            dst[((size_t)(b * HEADS + kh) * MSZ + rl) * HD + d] = f2bf(v);
        }
    }
}

// ---------------------------------------------------------------------------
// Kernel 2: attention per (b, head): Q = s (n rows), K = r, V = t (m rows).
// P[n,m] = Q.K/8, -inf where mask[m,n]==1, softmax over m, r2 = P@V.
// Two-pass over m (row max, then exp/sum/accumulate).
// ---------------------------------------------------------------------------
__global__ __launch_bounds__(256) void attn_kernel(
    const unsigned short* __restrict__ Qg,
    const unsigned short* __restrict__ Kg,
    const unsigned short* __restrict__ Vg,
    const unsigned short* __restrict__ Mp,   // packed mask bits
    unsigned short* __restrict__ R2)
{
    const int nb = blockIdx.x;
    const int kh = blockIdx.y;
    const int b  = blockIdx.z;
    const int n0 = nb * 128;
    const size_t off = (size_t)(b * HEADS + kh) * MSZ * HD;

    __shared__ alignas(16) unsigned short Qs[128 * 64];   // row-major [n][d]
    __shared__ alignas(16) unsigned short Ks[32 * 64];    // row-major [m][d]
    __shared__ alignas(16) unsigned short Vt[64 * 32];    // col-major [d][m]
    __shared__ alignas(16) unsigned short Ms16[32 * 8];   // bit mask [m][n/16]
    __shared__ alignas(16) unsigned short Ps[8][16 * 32]; // per-wave P staging

    const int tid  = threadIdx.x;
    const int wave = tid >> 5;
    const int lane = tid & 31;
    const int hf   = lane >> 4;
    const int l15  = lane & 15;
    const int wr0  = wave * 16;
    const float NEGINF = -__builtin_inff();

    #pragma unroll
    for (int it = 0; it < 4; ++it) {
        int g = tid + it * 256;
        *(uint4*)&Qs[g * 8] = *(const uint4*)&Qg[off + (size_t)n0 * HD + g * 8];
    }
    __syncthreads();

    Frag16 qa[2];
    #pragma unroll
    for (int c = 0; c < 2; ++c)
        ldfragA(qa[c], &Qs[(wr0 + l15) * 64 + c * 32], hf);

    float rmax[8], rsum[8];
    #pragma unroll
    for (int j = 0; j < 8; ++j) { rmax[j] = NEGINF; rsum[j] = 0.0f; }

    const int nw  = (wr0 + 8 * hf) >> 4;   // mask halfword index for this lane
    const int nsh = (wr0 + 8 * hf) & 15;   // bit shift base

    // ---------------- pass 1: row maxima ----------------
    for (int mb = 0; mb < MSZ; mb += 32) {
        __syncthreads();
        *(uint4*)&Ks[tid * 8] = *(const uint4*)&Kg[off + (size_t)mb * HD + tid * 8];
        Ms16[tid & 255] = Mp[(size_t)(mb + (tid >> 3)) * 64 + nb * 8 + (tid & 7)];
        __syncthreads();

        #pragma unroll
        for (int mt = 0; mt < 2; ++mt) {
            v8f s = {};
            #pragma unroll
            for (int c = 0; c < 2; ++c) {
                Frag16 kb;
                ldfragB(kb, &Ks[(mt * 16 + l15) * 64 + c * 32], hf);
                s = wmma_bf16(qa[c].v, kb.v, s);
            }
            const int ml = mt * 16 + l15;
            const unsigned mbits = Ms16[ml * 8 + nw];
            #pragma unroll
            for (int j = 0; j < 8; ++j) {
                float v = s[j] * 0.125f;
                if ((mbits >> (nsh + j)) & 1u) v = NEGINF;
                rmax[j] = fmaxf(rmax[j], v);
            }
        }
    }
    #pragma unroll
    for (int j = 0; j < 8; ++j)
        for (int d = 1; d < 16; d <<= 1)
            rmax[j] = fmaxf(rmax[j], __shfl_xor(rmax[j], d, 32));

    // ---------------- pass 2: exp, sum, P @ V ----------------
    v8f o[4] = {};
    for (int mb = 0; mb < MSZ; mb += 32) {
        __syncthreads();
        *(uint4*)&Ks[tid * 8] = *(const uint4*)&Kg[off + (size_t)mb * HD + tid * 8];
        Ms16[tid & 255] = Mp[(size_t)(mb + (tid >> 3)) * 64 + nb * 8 + (tid & 7)];
        // V transposed: Vt[d][m] (4 scalar u16 loads + 1 b64 store)
        #pragma unroll
        for (int it = 0; it < 2; ++it) {
            int g = tid + it * 256;
            int d = g & 63, mq = g >> 6;
            const unsigned short* vp = &Vg[off + (size_t)(mb + mq * 4) * HD + d];
            union { unsigned short s[4]; uint2 v; } t;
            t.s[0] = vp[0]; t.s[1] = vp[HD]; t.s[2] = vp[2 * HD]; t.s[3] = vp[3 * HD];
            *(uint2*)&Vt[d * 32 + mq * 4] = t.v;
        }
        __syncthreads();

        #pragma unroll
        for (int mt = 0; mt < 2; ++mt) {
            v8f s = {};
            #pragma unroll
            for (int c = 0; c < 2; ++c) {
                Frag16 kb;
                ldfragB(kb, &Ks[(mt * 16 + l15) * 64 + c * 32], hf);
                s = wmma_bf16(qa[c].v, kb.v, s);
            }
            const int ml = mt * 16 + l15;
            const unsigned mbits = Ms16[ml * 8 + nw];
            #pragma unroll
            for (int j = 0; j < 8; ++j) {
                float v = s[j] * 0.125f;
                if ((mbits >> (nsh + j)) & 1u) v = NEGINF;
                float pv = __expf(v - rmax[j]);
                rsum[j] += pv;
                Ps[wave][(j + 8 * hf) * 32 + mt * 16 + l15] = f2bf(pv);
            }
        }
        __syncthreads();

        Frag16 pa;
        ldfragA(pa, &Ps[wave][l15 * 32], hf);
        #pragma unroll
        for (int dt = 0; dt < 4; ++dt) {
            Frag16 vb;
            ldfragB(vb, &Vt[(dt * 16 + l15) * 32], hf);
            o[dt] = wmma_bf16(pa.v, vb.v, o[dt]);
        }
    }
    #pragma unroll
    for (int j = 0; j < 8; ++j)
        for (int d = 1; d < 16; d <<= 1)
            rsum[j] += __shfl_xor(rsum[j], d, 32);

    #pragma unroll
    for (int dt = 0; dt < 4; ++dt) {
        #pragma unroll
        for (int j = 0; j < 8; ++j) {
            int gn = n0 + wr0 + j + 8 * hf;
            int d  = dt * 16 + l15;
            R2[((size_t)(b * NSZ + gn)) * DIMSZ + kh * HD + d] = f2bf(o[dt][j] / rsum[j]);
        }
    }
}

// ---------------------------------------------------------------------------
// Kernel 3: out[b,n,q] = sum_kc r2[b,n,kc] * theta2[kc/64][q][kc%64]
// ---------------------------------------------------------------------------
__global__ __launch_bounds__(256) void outproj_kernel(
    const unsigned short* __restrict__ R2,
    const float* __restrict__ T2,
    float* __restrict__ out)
{
    const int rb = blockIdx.x;
    const int qt = blockIdx.y;
    const int row0 = rb * 128;
    const int q0   = qt * 64;

    __shared__ alignas(16) unsigned short Ab[128 * 32];
    __shared__ alignas(16) unsigned short Bt[64 * 32];   // col-major [q][kc]

    const int tid  = threadIdx.x;
    const int wave = tid >> 5;
    const int lane = tid & 31;
    const int hf   = lane >> 4;
    const int l15  = lane & 15;
    const int wr0  = wave * 16;

    v8f acc[4] = {};

    for (int c0 = 0; c0 < DIMSZ; c0 += 32) {
        __syncthreads();
        #pragma unroll
        for (int it = 0; it < 2; ++it) {
            int g = tid + it * 256;
            *(uint4*)&Ab[g * 8] =
                *(const uint4*)&R2[(size_t)(row0 + (g >> 2)) * DIMSZ + c0 + (g & 3) * 8];
        }
        // Bt[q][k] <- theta2[head][q0+q][d]; d contiguous -> float4 load
        #pragma unroll
        for (int it = 0; it < 2; ++it) {
            int g = tid + it * 256;
            int q = g & 63, kq = g >> 6;
            float4 f = *(const float4*)
                &T2[((size_t)(c0 >> 6) * DIMSZ + q0 + q) * HD + (c0 & 63) + kq * 4];
            st4bf(&Bt[q * 32 + kq * 4], f.x, f.y, f.z, f.w);
        }
        __syncthreads();

        Frag16 a;
        ldfragA(a, &Ab[(wr0 + l15) * 32], hf);
        #pragma unroll
        for (int t = 0; t < 4; ++t) {
            Frag16 b;
            ldfragB(b, &Bt[(t * 16 + l15) * 32], hf);
            acc[t] = wmma_bf16(a.v, b.v, acc[t]);
        }
    }

    #pragma unroll
    for (int t = 0; t < 4; ++t) {
        #pragma unroll
        for (int j = 0; j < 8; ++j) {
            int g = row0 + wr0 + j + 8 * hf;
            out[(size_t)g * DIMSZ + q0 + t * 16 + l15] = acc[t][j];
        }
    }
}

// ---------------------------------------------------------------------------
// Launch
// ---------------------------------------------------------------------------
extern "C" void kernel_launch(void* const* d_in, const int* in_sizes, int n_in,
                              void* d_out, int out_size, void* d_ws, size_t ws_size,
                              hipStream_t stream) {
    const float* x  = (const float*)d_in[0];
    const float* y  = (const float*)d_in[1];
    const int*   mk = (const int*)  d_in[2];
    const float* l1 = (const float*)d_in[3];
    const float* l2 = (const float*)d_in[4];
    const float* t1 = (const float*)d_in[5];
    const float* t2 = (const float*)d_in[6];
    const float* bl = (const float*)d_in[7];
    const float* bt = (const float*)d_in[8];
    float* out = (float*)d_out;

    const size_t PROJ_ELEMS = (size_t)BSZ * HEADS * MSZ * HD;   // 4M bf16 each
    unsigned short* wr  = (unsigned short*)d_ws;                // r = x@lambda1
    unsigned short* ws  = wr  + PROJ_ELEMS;                     // s = y@lambda2
    unsigned short* wt  = ws  + PROJ_ELEMS;                     // t = x@theta1
    unsigned short* wr2 = wt  + PROJ_ELEMS;                     // r2 [b][n][512]
    unsigned short* Mp  = wr2 + (size_t)BSZ * NSZ * DIMSZ;      // packed mask

    pack_mask_kernel<<<dim3((MSZ * NSZ / 16) / 256), 256, 0, stream>>>(mk, Mp);
    proj_kernel<<<dim3(ROWS / 128, HEADS, 3), 256, 0, stream>>>(
        x, y, l1, l2, t1, bl, bt, wr, ws, wt);
    attn_kernel<<<dim3(NSZ / 128, HEADS, BSZ), 256, 0, stream>>>(
        ws, wr, wt, Mp, wr2);
    outproj_kernel<<<dim3(ROWS / 128, DIMSZ / 64), 256, 0, stream>>>(
        wr2, t2, out);
}
